// GINE_13898514170649
// MI455X (gfx1250) — matrix operand order repaired and checked by reading, not compiled, not run on hardware
//
#include <hip/hip_runtime.h>
#include <hip/hip_bf16.h>

// ---------------------------------------------------------------------------
// GINE GNN forward for MI455X (gfx1250, wave32).
//  - Dense layers on V_WMMA_F32_16X16X32_BF16 (bf16 in, f32 accum), with
//    bias + LayerNorm(+LayerNorm) + ReLU + residual fused in the epilogue.
//  - GEMM: K/OUT compile-time, LDS double-buffered (1 barrier per k-step),
//    2 column tiles per wave (shared A fragment -> 3 ds_load_b128 per wmma),
//    A staged as [16][16 k-pairs] u32 (pad 20), B staged TRANSPOSED as
//    [OUT][16 k-pairs] u32 (pad 20) so fragments are contiguous b128 loads.
//  - Edge message passing: fused gather/edge-linear/relu/scatter-atomic;
//    h (51 MB) is L2-resident (192 MB L2) so this phase is L2-bound.
// ---------------------------------------------------------------------------

typedef __attribute__((ext_vector_type(16))) __bf16 bf16x16;
typedef __attribute__((ext_vector_type(8)))  float  floatx8;

#define HN 256   // hidden size H
#define GN 1024  // number of graphs G
#define TN 12    // classes

__device__ __forceinline__ float wave_allreduce_sum(float v) {
#pragma unroll
  for (int m = 16; m >= 1; m >>= 1) v += __shfl_xor(v, m, 32);
  return v;
}

__device__ __forceinline__ unsigned pack_bf16(float a, float b) {
  union { __bf16 h[2]; unsigned u; } p;
  p.h[0] = (__bf16)a;
  p.h[1] = (__bf16)b;
  return p.u;
}

// ---------------------------------------------------------------------------
// Fused GEMM:  out[rows,OUT] = post( X[rows,K] @ W[K,OUT] + bias )  (+resid)
//   LN_MODE 0: ReLU   1: ReLU(LN(.))   2: ReLU(LN(ReLU(LN(.))))
// rows must be a multiple of 16 (true for all call sites).
// Workgroup: 16 rows x OUT cols; OUT/32 waves, each wave owns a 16x32 slab
// (two 16x16 WMMA tiles sharing one A fragment).
// ---------------------------------------------------------------------------
template <int LN_MODE, int K, int OUT>
__global__ void __launch_bounds__(OUT)
wmma_gemm_ln_relu(const float* __restrict__ X, const float* __restrict__ W,
                  const float* __restrict__ bias,
                  const float* __restrict__ g1v, const float* __restrict__ b1v,
                  const float* __restrict__ g2v, const float* __restrict__ b2v,
                  const float* __restrict__ resid, float* __restrict__ out) {
  constexpr int NT = OUT;              // threads
  constexpr int NW = OUT / 32;         // waves (one 16x32 slab each)
  constexpr int NK = K / 32;           // k-steps
  constexpr int APAD = 20;             // u32 stride per row (pad: 16 -> 20)
  constexpr int ASZ = 16 * APAD;       // A buffer u32s
  constexpr int BSZ = OUT * APAD;      // B buffer u32s

  extern __shared__ char smem[];
  unsigned* As0 = (unsigned*)smem;       // [2][16][APAD]
  unsigned* Bs0 = As0 + 2 * ASZ;         // [2][OUT][APAD]
  float* tile = (float*)smem;            // epilogue alias: [16][OUT]

  const int tid = threadIdx.x;
  const int lane = tid & 31;
  const int wid = tid >> 5;
  const int row0 = blockIdx.x * 16;
  const int half = lane >> 4;
  const int l16 = lane & 15;
  const int colbase = wid * 32;

  // B first (bulk loads issue early), A second with replicated loads so the
  // load stream has no divergent branch before it.
  auto stageB = [&](int t, int b) {
    const float* Wp = W + (size_t)t * 32 * OUT;
    unsigned* dst = Bs0 + b * BSZ;
#pragma unroll
    for (int i = tid; i < OUT * 4; i += NT) {    // 16 k-pairs x OUT/4 col-quads
      int kp = i / (OUT / 4);
      int c4 = (i - kp * (OUT / 4)) * 4;
      float4 r0 = *(const float4*)(Wp + (size_t)(2 * kp) * OUT + c4);
      float4 r1 = *(const float4*)(Wp + (size_t)(2 * kp + 1) * OUT + c4);
      unsigned* p = dst + c4 * APAD + kp;        // transposed: [col][k-pair]
      p[0 * APAD] = pack_bf16(r0.x, r1.x);
      p[1 * APAD] = pack_bf16(r0.y, r1.y);
      p[2 * APAD] = pack_bf16(r0.z, r1.z);
      p[3 * APAD] = pack_bf16(r0.w, r1.w);
    }
  };
  auto stageA = [&](int t, int b) {
    const float* Xp = X + (size_t)row0 * K + t * 32;
    unsigned* dst = As0 + b * ASZ;
    int i = tid & 127;                           // replicate: load unconditional
    int r = i >> 3, q = i & 7;                   // 16 rows x 8 float4-quads
    float4 v = *(const float4*)(Xp + (size_t)r * K + q * 4);
    if (tid < 128) {
      unsigned* p = dst + r * APAD + q * 2;
      p[0] = pack_bf16(v.x, v.y);
      p[1] = pack_bf16(v.z, v.w);
    }
  };

  floatx8 acc0 = {0.f, 0.f, 0.f, 0.f, 0.f, 0.f, 0.f, 0.f};
  floatx8 acc1 = {0.f, 0.f, 0.f, 0.f, 0.f, 0.f, 0.f, 0.f};
  union Frag { uint4 q[2]; bf16x16 v; };

  stageB(0, 0);
  stageA(0, 0);
#pragma unroll 2
  for (int t = 0; t < NK; ++t) {
    __syncthreads();
    if (t + 1 < NK) {  // prefetch next k-step into the other buffer
      stageB(t + 1, (t + 1) & 1);
      stageA(t + 1, (t + 1) & 1);
    }
    const int b = t & 1;
    // A fragment: lane l16 = row; j<8 -> k=half*8+j, j>=8 -> k=16+half*8+(j-8)
    Frag af, bf0, bf1;
    const unsigned* ap = As0 + b * ASZ + l16 * APAD + half * 4;
    af.q[0] = *(const uint4*)(ap);
    af.q[1] = *(const uint4*)(ap + 8);
    // B fragments: lane l16 = col within tile; j<8 -> k=half*8+j, j>=8 -> k=16+half*8+(j-8)
    const unsigned* bp = Bs0 + b * BSZ + (colbase + l16) * APAD + half * 4;
    bf0.q[0] = *(const uint4*)(bp);
    bf0.q[1] = *(const uint4*)(bp + 8);
    bf1.q[0] = *(const uint4*)(bp + 16 * APAD);
    bf1.q[1] = *(const uint4*)(bp + 16 * APAD + 8);
    acc0 = __builtin_amdgcn_wmma_f32_16x16x32_bf16(false, af.v, false, bf0.v,
                                                   (short)0, acc0, false, false);
    acc1 = __builtin_amdgcn_wmma_f32_16x16x32_bf16(false, af.v, false, bf1.v,
                                                   (short)0, acc1, false, false);
  }
  __syncthreads();  // staging buffers dead; reuse LDS as f32 tile

  // Spill accumulators (+bias). C/D layout: acc[i] is (m = i + 8*half, n = l16).
  {
    float bc0 = bias ? bias[colbase + l16] : 0.f;
    float bc1 = bias ? bias[colbase + 16 + l16] : 0.f;
#pragma unroll
    for (int i = 0; i < 8; ++i) {
      int m = i + half * 8;
      tile[m * OUT + colbase + l16] = acc0[i] + bc0;
      tile[m * OUT + colbase + 16 + l16] = acc1[i] + bc1;
    }
  }
  __syncthreads();

  if (LN_MODE == 0) {
#pragma unroll
    for (int i = tid; i < 16 * OUT; i += NT) tile[i] = fmaxf(tile[i], 0.f);
    __syncthreads();
  } else {
#pragma unroll
    for (int pass = 0; pass < LN_MODE; ++pass) {
      const float* g = (pass == 0) ? g1v : g2v;
      const float* b = (pass == 0) ? b1v : b2v;
      for (int r = wid; r < 16; r += NW) {  // one wave owns a row
        float s = 0.f, ss = 0.f;
#pragma unroll
        for (int c = lane * 4; c < OUT; c += 128) {
          float4 v = *(const float4*)&tile[r * OUT + c];
          s += v.x + v.y + v.z + v.w;
          ss += v.x * v.x + v.y * v.y + v.z * v.z + v.w * v.w;
        }
        s = wave_allreduce_sum(s);
        ss = wave_allreduce_sum(ss);
        float mean = s / (float)OUT;
        float var = ss / (float)OUT - mean * mean;
        float rstd = rsqrtf(var + 1e-5f);
#pragma unroll
        for (int c = lane * 4; c < OUT; c += 128) {
          float4 v = *(const float4*)&tile[r * OUT + c];
          float4 gv = *(const float4*)&g[c];
          float4 bv = *(const float4*)&b[c];
          v.x = fmaxf((v.x - mean) * rstd * gv.x + bv.x, 0.f);
          v.y = fmaxf((v.y - mean) * rstd * gv.y + bv.y, 0.f);
          v.z = fmaxf((v.z - mean) * rstd * gv.z + bv.z, 0.f);
          v.w = fmaxf((v.w - mean) * rstd * gv.w + bv.w, 0.f);
          *(float4*)&tile[r * OUT + c] = v;
        }
      }
      __syncthreads();
    }
  }

  // Vectorized store (+residual).
#pragma unroll
  for (int i = tid; i < 16 * (OUT / 4); i += NT) {
    int r = i / (OUT / 4);
    int c4 = (i - r * (OUT / 4)) * 4;
    float4 v = *(const float4*)&tile[r * OUT + c4];
    size_t off = (size_t)(row0 + r) * OUT + c4;
    if (resid) {
      float4 rv = *(const float4*)&resid[off];
      v.x += rv.x; v.y += rv.y; v.z += rv.z; v.w += rv.w;
    }
    *(float4*)&out[off] = v;
  }
}

// ---------------------------------------------------------------------------
// Layer-0 first MLP: z0[N,9] @ W[9,256] + b, LayerNorm, ReLU. K=9 -> no WMMA.
// ---------------------------------------------------------------------------
__global__ void dense9_ln_relu(const float* __restrict__ Z,
                               const float* __restrict__ W,
                               const float* __restrict__ bias,
                               const float* __restrict__ g,
                               const float* __restrict__ b,
                               float* __restrict__ out, int rows) {
  __shared__ float zr[9];
  __shared__ float red[16];
  int n = blockIdx.x;
  int c = threadIdx.x;  // 256
  if (c < 9) zr[c] = Z[(size_t)n * 9 + c];
  __syncthreads();
  float acc = bias[c];
#pragma unroll
  for (int k = 0; k < 9; ++k) acc += zr[k] * W[k * HN + c];
  float s = wave_allreduce_sum(acc);
  float ss = wave_allreduce_sum(acc * acc);
  if ((c & 31) == 0) { red[c >> 5] = s; red[8 + (c >> 5)] = ss; }
  __syncthreads();
  float tot = 0.f, tot2 = 0.f;
#pragma unroll
  for (int i = 0; i < 8; ++i) { tot += red[i]; tot2 += red[8 + i]; }
  float mean = tot / (float)HN;
  float var = tot2 / (float)HN - mean * mean;
  float rstd = rsqrtf(var + 1e-5f);
  out[(size_t)n * HN + c] = fmaxf((acc - mean) * rstd * g[c] + b[c], 0.f);
}

// ---------------------------------------------------------------------------
// Fused edge message: agg[dst] += relu(h[src] + edge_attr@We + be). H=256.
// One 64-thread block per edge, float4 per thread, scalar f32 atomics.
// ---------------------------------------------------------------------------
__global__ void edge_msg_h(const float* __restrict__ h,
                           const float* __restrict__ ea,
                           const float* __restrict__ We,  // [3,256]
                           const float* __restrict__ be,
                           const int* __restrict__ src, const int* __restrict__ dst,
                           float* __restrict__ agg, int nE) {
  int e = blockIdx.x;
  if (e >= nE) return;
  int s = src[e], d = dst[e];
  float a0 = ea[e * 3 + 0], a1 = ea[e * 3 + 1], a2 = ea[e * 3 + 2];
  int c = threadIdx.x * 4;
  float4 hv = *(const float4*)(h + (size_t)s * HN + c);
  float4 w0 = *(const float4*)(We + 0 * HN + c);
  float4 w1 = *(const float4*)(We + 1 * HN + c);
  float4 w2 = *(const float4*)(We + 2 * HN + c);
  float4 bv = *(const float4*)(be + c);
  float m0 = fmaxf(hv.x + a0 * w0.x + a1 * w1.x + a2 * w2.x + bv.x, 0.f);
  float m1 = fmaxf(hv.y + a0 * w0.y + a1 * w1.y + a2 * w2.y + bv.y, 0.f);
  float m2 = fmaxf(hv.z + a0 * w0.z + a1 * w1.z + a2 * w2.z + bv.z, 0.f);
  float m3 = fmaxf(hv.w + a0 * w0.w + a1 * w1.w + a2 * w2.w + bv.w, 0.f);
  float* p = agg + (size_t)d * HN + c;
  atomicAdd(p + 0, m0); atomicAdd(p + 1, m1);
  atomicAdd(p + 2, m2); atomicAdd(p + 3, m3);
}

// Layer-0 variant, fi = 9.
__global__ void edge_msg9(const float* __restrict__ x, const float* __restrict__ ea,
                          const float* __restrict__ We,  // [3,9]
                          const float* __restrict__ be,
                          const int* __restrict__ src, const int* __restrict__ dst,
                          float* __restrict__ agg, int nE) {
  int e = blockIdx.x;
  int t = threadIdx.x;
  if (e >= nE || t >= 9) return;
  int s = src[e], d = dst[e];
  float a0 = ea[e * 3 + 0], a1 = ea[e * 3 + 1], a2 = ea[e * 3 + 2];
  float m = x[(size_t)s * 9 + t] + a0 * We[t] + a1 * We[9 + t] + a2 * We[18 + t] + be[t];
  atomicAdd(agg + (size_t)d * 9 + t, fmaxf(m, 0.f));
}

__global__ void add_vn(float* __restrict__ h, const float* __restrict__ vn,
                       const int* __restrict__ batch, int n_nodes) {
  size_t i = (size_t)blockIdx.x * 256 + threadIdx.x;
  if (i >= (size_t)n_nodes * HN) return;
  int n = (int)(i >> 8), c = (int)(i & 255);
  h[i] += vn[(size_t)batch[n] * HN + c];
}

// dstbuf[batch[n]] += h[n]  (used for vn aggregation and sum-pool)
__global__ void seg_accum(const float* __restrict__ h, const int* __restrict__ batch,
                          float* __restrict__ dstbuf, int n_nodes) {
  int n = blockIdx.x;
  if (n >= n_nodes) return;
  int c = threadIdx.x * 4;
  int g = batch[n];
  float4 v = *(const float4*)(h + (size_t)n * HN + c);
  float* p = dstbuf + (size_t)g * HN + c;
  atomicAdd(p + 0, v.x); atomicAdd(p + 1, v.y);
  atomicAdd(p + 2, v.z); atomicAdd(p + 3, v.w);
}

// max-pool: h >= 0 after relu, so nonneg float bits are monotone as uint.
__global__ void seg_max_u(const float* __restrict__ h, const int* __restrict__ batch,
                          unsigned* __restrict__ dstmax, int n_nodes) {
  int n = blockIdx.x;
  if (n >= n_nodes) return;
  int c = threadIdx.x;
  int g = batch[n];
  atomicMax(dstmax + (size_t)g * HN + c, __float_as_uint(h[(size_t)n * HN + c]));
}

__global__ void count_nodes(const int* __restrict__ batch, float* __restrict__ counts,
                            int n_nodes) {
  int n = blockIdx.x * 256 + threadIdx.x;
  if (n < n_nodes) atomicAdd(counts + batch[n], 1.f);
}

__global__ void bcast_vn(const float* __restrict__ emb, float* __restrict__ vn) {
  vn[(size_t)blockIdx.x * HN + threadIdx.x] = emb[threadIdx.x];
}

__global__ void fill_u32(unsigned* __restrict__ p, unsigned v, int n) {
  int i = blockIdx.x * 256 + threadIdx.x;
  if (i < n) p[i] = v;
}

// gate[n] = t[n,:] . W2 + b2 ; atomicMax of order-mapped bits into gmax[batch].
__global__ void gate_score(const float* __restrict__ t, const float* __restrict__ W2,
                           const float* __restrict__ b2, const int* __restrict__ batch,
                           float* __restrict__ gate, unsigned* __restrict__ gmax,
                           int n_nodes) {
  __shared__ float red[8];
  int n = blockIdx.x;
  if (n >= n_nodes) return;
  int c = threadIdx.x;
  float v = t[(size_t)n * HN + c] * W2[c];
  v = wave_allreduce_sum(v);
  if ((c & 31) == 0) red[c >> 5] = v;
  __syncthreads();
  if (c == 0) {
    float sum = b2[0];
#pragma unroll
    for (int i = 0; i < 8; ++i) sum += red[i];
    gate[n] = sum;
    unsigned bits = __float_as_uint(sum);
    unsigned m = (bits & 0x80000000u) ? ~bits : (bits | 0x80000000u);
    atomicMax(gmax + batch[n], m);
  }
}

__global__ void gate_exp(const float* __restrict__ gate, const unsigned* __restrict__ gmax,
                         const int* __restrict__ batch, float* __restrict__ esc,
                         float* __restrict__ denom, int n_nodes) {
  int n = blockIdx.x * 256 + threadIdx.x;
  if (n >= n_nodes) return;
  unsigned u = gmax[batch[n]];
  unsigned bits = (u & 0x80000000u) ? (u & 0x7fffffffu) : ~u;
  float e = __expf(gate[n] - __uint_as_float(bits));
  esc[n] = e;
  atomicAdd(denom + batch[n], e);
}

__global__ void att_accum(const float* __restrict__ h, const float* __restrict__ esc,
                          const float* __restrict__ denom, const int* __restrict__ batch,
                          float* __restrict__ s_att, int n_nodes) {
  int n = blockIdx.x;
  if (n >= n_nodes) return;
  int c = threadIdx.x * 4;
  int g = batch[n];
  float coef = esc[n] / denom[g];
  float4 v = *(const float4*)(h + (size_t)n * HN + c);
  float* p = s_att + (size_t)g * HN + c;
  atomicAdd(p + 0, coef * v.x); atomicAdd(p + 1, coef * v.y);
  atomicAdd(p + 2, coef * v.z); atomicAdd(p + 3, coef * v.w);
}

__global__ void build_xp(const float* __restrict__ s_add, const unsigned* __restrict__ s_maxu,
                         const float* __restrict__ s_att, const float* __restrict__ counts,
                         float* __restrict__ xp) {
  int g = blockIdx.x;
  int c = threadIdx.x;  // 256
  float cnt = fmaxf(counts[g], 1.f);
  float sa = s_add[(size_t)g * HN + c];
  size_t base = (size_t)g * 4 * HN;
  xp[base + c] = sa / cnt;
  xp[base + HN + c] = sa;
  xp[base + 2 * HN + c] = __uint_as_float(s_maxu[(size_t)g * HN + c]);
  xp[base + 3 * HN + c] = s_att[(size_t)g * HN + c];
}

__global__ void clf_out(const float* __restrict__ z, const float* __restrict__ W3,
                        const float* __restrict__ b3, float* __restrict__ out) {
  int g = blockIdx.x;
  int t = threadIdx.x;
  if (t >= TN) return;
  float acc = b3[t];
  for (int c = 0; c < HN; ++c) acc += z[(size_t)g * HN + c] * W3[c * TN + t];
  out[(size_t)g * TN + t] = acc;
}

// ---------------------------------------------------------------------------
extern "C" void kernel_launch(void* const* d_in, const int* in_sizes, int n_in,
                              void* d_out, int out_size, void* d_ws, size_t ws_size,
                              hipStream_t stream) {
  const int N_ = in_sizes[0] / 9;   // 50000
  const int E_ = in_sizes[1] / 3;   // 800000

  // ---- input mapping: top-level dict insertion order; params flattened via
  //      jax tree_leaves (sorted dict keys, list order preserved):
  //  0:x  1:edge_attr
  //  2..11  clf:  W1,W2,W3,b1,b2,b3,bb1,bb2,g1,g2
  //  12+12l conv[l]: W1,W2,We,b1,b2,bb1,bb2,be,bn,g1,g2,gn
  //  72..75 gate: W1,W2,b1,b2
  //  76     vn_emb
  //  77+8l  vn_mlp[l]: W1,W2,b1,b2,bb1,bb2,g1,g2
  //  109:edge_index  110:batch
  const float* x  = (const float*)d_in[0];
  const float* ea = (const float*)d_in[1];
  const float* cW1 = (const float*)d_in[2];
  const float* cW2 = (const float*)d_in[3];
  const float* cW3 = (const float*)d_in[4];
  const float* cb1 = (const float*)d_in[5];
  const float* cb2 = (const float*)d_in[6];
  const float* cb3 = (const float*)d_in[7];
  const float* cbb1 = (const float*)d_in[8];
  const float* cbb2 = (const float*)d_in[9];
  const float* cg1 = (const float*)d_in[10];
  const float* cg2 = (const float*)d_in[11];
  const float* conv[5][12];
  for (int l = 0; l < 5; ++l)
    for (int j = 0; j < 12; ++j) conv[l][j] = (const float*)d_in[12 + 12 * l + j];
  // conv[l]: 0:W1 1:W2 2:We 3:b1 4:b2 5:bb1 6:bb2 7:be 8:bn 9:g1 10:g2 11:gn
  const float* gW1 = (const float*)d_in[72];
  const float* gW2 = (const float*)d_in[73];
  const float* gb1 = (const float*)d_in[74];
  const float* gb2 = (const float*)d_in[75];
  const float* vn_emb = (const float*)d_in[76];
  const float* vmlp[4][8];
  for (int l = 0; l < 4; ++l)
    for (int j = 0; j < 8; ++j) vmlp[l][j] = (const float*)d_in[77 + 8 * l + j];
  // vmlp[l]: 0:W1 1:W2 2:b1 3:b2 4:bb1 5:bb2 6:g1 7:g2
  const int* eidx  = (const int*)d_in[109];
  const int* batch = (const int*)d_in[110];
  const int* src = eidx;
  const int* dst = eidx + E_;

  // ---- workspace partition
  char* w = (char*)d_ws;
  auto take = [&](size_t bytes) -> char* {
    char* p = w;
    w += (bytes + 255) & ~(size_t)255;
    return p;
  };
  float* A    = (float*)take((size_t)N_ * HN * 4);  // h
  float* Bbuf = (float*)take((size_t)N_ * HN * 4);  // agg / gate-t scratch
  float* Cbuf = (float*)take((size_t)N_ * HN * 4);  // MLP intermediate
  float* vn   = (float*)take((size_t)GN * HN * 4);
  float* ubuf = (float*)take((size_t)GN * HN * 4);
  float* vtmp = (float*)take((size_t)GN * HN * 4);
  float* s_add = (float*)take((size_t)GN * HN * 4);
  unsigned* s_maxu = (unsigned*)take((size_t)GN * HN * 4);
  float* s_att = (float*)take((size_t)GN * HN * 4);
  float* counts = (float*)take((size_t)GN * 4);
  float* gatev  = (float*)take((size_t)N_ * 4);
  float* escv   = (float*)take((size_t)N_ * 4);
  unsigned* gmaxu = (unsigned*)take((size_t)GN * 4);
  float* denom  = (float*)take((size_t)GN * 4);
  float* xp = (float*)take((size_t)GN * 4 * HN * 4);
  float* c1 = (float*)take((size_t)GN * 2 * HN * 4);
  float* c2 = (float*)take((size_t)GN * HN * 4);

  const int rowsBlkN = N_ / 16;     // N_ = 50000 = 3125*16
  const int rowsBlkG = GN / 16;
  // dynamic LDS: 2 k-step buffers of A[16][20]u32 + B[OUT][20]u32 (tile aliases)
  const size_t sm256 = (size_t)2 * (16 + 256) * 20 * 4;  // 43,520 B
  const size_t sm512 = (size_t)2 * (16 + 512) * 20 * 4;  // 84,480 B
  const dim3 blk256(256);   // OUT=256: 8 waves
  const dim3 blk512(512);   // OUT=512: 16 waves

  // ---- virtual node init
  bcast_vn<<<GN, HN, 0, stream>>>(vn_emb, vn);

  // ---- layer 0 (fi = 9)
  hipMemcpyAsync(Bbuf, x, (size_t)N_ * 9 * 4, hipMemcpyDeviceToDevice, stream);
  edge_msg9<<<E_, 16, 0, stream>>>(x, ea, conv[0][2], conv[0][7], src, dst, Bbuf, E_);
  dense9_ln_relu<<<N_, HN, 0, stream>>>(Bbuf, conv[0][0], conv[0][3], conv[0][9],
                                        conv[0][5], Cbuf, N_);
  wmma_gemm_ln_relu<2, 256, 256><<<rowsBlkN, blk256, sm256, stream>>>(
      Cbuf, conv[0][1], conv[0][4], conv[0][10], conv[0][6], conv[0][11], conv[0][8],
      nullptr, A);

  for (int l = 0; l < 5; ++l) {
    if (l > 0) {
      add_vn<<<N_, HN, 0, stream>>>(A, vn, batch, N_);
      hipMemcpyAsync(Bbuf, A, (size_t)N_ * HN * 4, hipMemcpyDeviceToDevice, stream);
      edge_msg_h<<<E_, 64, 0, stream>>>(A, ea, conv[l][2], conv[l][7], src, dst, Bbuf, E_);
      wmma_gemm_ln_relu<1, 256, 256><<<rowsBlkN, blk256, sm256, stream>>>(
          Bbuf, conv[l][0], conv[l][3], conv[l][9], conv[l][5], nullptr, nullptr,
          nullptr, Cbuf);
      wmma_gemm_ln_relu<2, 256, 256><<<rowsBlkN, blk256, sm256, stream>>>(
          Cbuf, conv[l][1], conv[l][4], conv[l][10], conv[l][6], conv[l][11], conv[l][8],
          A /*residual*/, A);
    }
    if (l < 4) {  // virtual-node update: vn = MLP(vn + segsum(h))
      hipMemcpyAsync(ubuf, vn, (size_t)GN * HN * 4, hipMemcpyDeviceToDevice, stream);
      seg_accum<<<N_, 64, 0, stream>>>(A, batch, ubuf, N_);
      wmma_gemm_ln_relu<1, 256, 256><<<rowsBlkG, blk256, sm256, stream>>>(
          ubuf, vmlp[l][0], vmlp[l][2], vmlp[l][6], vmlp[l][4], nullptr, nullptr,
          nullptr, vtmp);
      wmma_gemm_ln_relu<1, 256, 256><<<rowsBlkG, blk256, sm256, stream>>>(
          vtmp, vmlp[l][1], vmlp[l][3], vmlp[l][7], vmlp[l][5], nullptr, nullptr,
          nullptr, vn);
    }
  }

  // ---- global pooling
  hipMemsetAsync(s_add, 0, (size_t)GN * HN * 4, stream);
  hipMemsetAsync(s_maxu, 0, (size_t)GN * HN * 4, stream);  // h>=0: 0 == -inf proxy
  hipMemsetAsync(s_att, 0, (size_t)GN * HN * 4, stream);
  hipMemsetAsync(counts, 0, (size_t)GN * 4, stream);
  hipMemsetAsync(denom, 0, (size_t)GN * 4, stream);
  fill_u32<<<(GN + 255) / 256, 256, 0, stream>>>(gmaxu, 0x007FFFFFu, GN);  // map(-inf)

  seg_accum<<<N_, 64, 0, stream>>>(A, batch, s_add, N_);
  seg_max_u<<<N_, HN, 0, stream>>>(A, batch, s_maxu, N_);
  count_nodes<<<(N_ + 255) / 256, 256, 0, stream>>>(batch, counts, N_);

  // attention gate: t = relu(h @ gW1 + gb1) on WMMA, then score + segment softmax
  wmma_gemm_ln_relu<0, 256, 256><<<rowsBlkN, blk256, sm256, stream>>>(
      A, gW1, gb1, nullptr, nullptr, nullptr, nullptr, nullptr, Bbuf);
  gate_score<<<N_, HN, 0, stream>>>(Bbuf, gW2, gb2, batch, gatev, gmaxu, N_);
  gate_exp<<<(N_ + 255) / 256, 256, 0, stream>>>(gatev, gmaxu, batch, escv, denom, N_);
  att_accum<<<N_, 64, 0, stream>>>(A, escv, denom, batch, s_att, N_);

  build_xp<<<GN, HN, 0, stream>>>(s_add, s_maxu, s_att, counts, xp);

  // ---- classifier
  wmma_gemm_ln_relu<1, 1024, 512><<<rowsBlkG, blk512, sm512, stream>>>(
      xp, cW1, cb1, cg1, cbb1, nullptr, nullptr, nullptr, c1);
  wmma_gemm_ln_relu<1, 512, 256><<<rowsBlkG, blk256, sm256, stream>>>(
      c1, cW2, cb2, cg2, cbb2, nullptr, nullptr, nullptr, c2);
  clf_out<<<GN, 16, 0, stream>>>(c2, cW3, cb3, (float*)d_out);

  (void)n_in; (void)out_size; (void)ws_size;
}
